// GraphAnomalyDetectionModel_83056077570952
// MI455X (gfx1250) — matrix operand order repaired and checked by reading, not compiled, not run on hardware
//
#include <hip/hip_runtime.h>
#include <math.h>

// ---------------------------------------------------------------------------
// MI455X (gfx1250) HGT forward. Dense matmuls use v_wmma_f32_16x16x32_f16
// (wave32). LDS tiles are stored as f16 pre-swizzled into the exact WMMA
// fragment order (ISA 7.12.2), so each lane fetches its A/B fragments with a
// single contiguous 32B LDS read (ds_load_b128 x2) per matrix per K-step.
// The column guard is scalarized via readfirstlane so WMMA always executes
// with EXEC all-1s under a scalar branch. Edge softmax/aggregation uses f32
// atomics that stay L2-resident (all node feature tables < 192MB L2).
// ---------------------------------------------------------------------------

typedef __attribute__((ext_vector_type(16))) _Float16 v16h;
typedef __attribute__((ext_vector_type(2)))  _Float16 v2h;
typedef __attribute__((ext_vector_type(8)))  float    v8f;

#define NCN 100000
#define NMN 50000
#define NEE 500000

__device__ __forceinline__ float gelu_f(float x) {
  return 0.5f * x * (1.f + erff(x * 0.70710678118654752f));
}

__device__ __forceinline__ void atomicMaxF(float* addr, float val) {
  unsigned int* ai = (unsigned int*)addr;
  unsigned int cur = *ai;
  while (__uint_as_float(cur) < val) {
    unsigned int prev = atomicCAS(ai, cur, __float_as_uint(val));
    if (prev == cur) break;
    cur = prev;
  }
}

// Precompute sigmoid of the (scalar) skip gates once, so GEMM epilogues just
// load a ready blend factor instead of running an exp/div chain per thread.
__global__ void prep_gates_kernel(const float* __restrict__ g0,
                                  const float* __restrict__ g1,
                                  float* __restrict__ out) {
  if (threadIdx.x == 0) {
    out[0] = 1.f / (1.f + expf(-g0[0]));
    out[1] = 1.f / (1.f + expf(-g1[0]));
  }
}

// Generic GEMM: Y[M,N] = actY( X[M,K] @ W[K,N] + bias ), optional gelu on X,
// optional pre-sigmoided skip blend. One 16x16 WMMA tile per wave, 4 waves
// (64 output cols) per 128-thread block. K is a compile-time constant so the
// K-loop fully unrolls into back-to-back v_wmma ops.
template <int K>
__global__ void __launch_bounds__(128)
gemm_wmma_t(const float* __restrict__ X, int ldx,
            const float* __restrict__ Wt, int ldw,
            const float* __restrict__ bias,
            float* __restrict__ Y, int ldy,
            int M, int N,
            int geluX, int actY,
            const float* __restrict__ skipX, int ldskip,
            const float* __restrict__ betaPtr) {
  // Pre-swizzled f16 tiles, 32B-aligned so fragment reads are b128 pairs.
  __shared__ v16h sAv[32];    // A tile: [lane(32)][e(16)]   (16x32 fp16, 1KB)
  __shared__ v16h sBv[128];   // B tile: [wave(4)][lane(32)][e(16)] (32x64, 4KB)
  _Float16* sA = (_Float16*)sAv;
  _Float16* sB = (_Float16*)sBv;

  const int tid  = threadIdx.x;
  const int wave = tid >> 5;
  const int lane = tid & 31;
  const int half = lane >> 4;
  const int lr   = lane & 15;
  const int row0 = blockIdx.x * 16;
  // wave-uniform by construction; readfirstlane makes it an SGPR so the
  // guard compiles to s_cmp/s_cbranch and EXEC stays all-1s around WMMA.
  const int colW = __builtin_amdgcn_readfirstlane(blockIdx.y * 64 + wave * 16);

  v8f acc = {0.f, 0.f, 0.f, 0.f, 0.f, 0.f, 0.f, 0.f};

#pragma unroll
  for (int k0 = 0; k0 < K; k0 += 32) {
    // ---- stage A (16 rows x 32 k), paired along k, swizzled to fragments
    // pair p: row r = p>>4, kp = p&15 -> k = 2*kp(,+1)
    // dest: half=(kp>>2)&1, vgpr i=(kp&3)|((kp>>3)<<2), lane=r+16*half, e=2i
#pragma unroll
    for (int p = tid; p < 256; p += 128) {
      int r  = p >> 4;
      int kp = p & 15;
      int kk = k0 + 2 * kp;
      float v0 = 0.f, v1 = 0.f;
      if (kk < K) {
        const float* xp = X + (size_t)(row0 + r) * ldx + kk;
        v0 = xp[0];
        v1 = xp[1];
        if (geluX) { v0 = gelu_f(v0); v1 = gelu_f(v1); }
      }
      int h  = (kp >> 2) & 1;
      int i  = (kp & 3) | ((kp >> 3) << 2);
      int dl = r + 16 * h;
      *(v2h*)&sA[dl * 16 + 2 * i] = (v2h){(_Float16)v0, (_Float16)v1};
    }
    // ---- stage B (32 k x 64 cols), paired along k, swizzled to fragments
    // pair p: kp = p>>6 -> k = 2*kp(,+1), col c = p&63
    // dest: wave=c>>4, lane=(c&15)+16*((k>>3)&1), e=(k&7)|((k>>4)<<3) (pair stays in-half)
#pragma unroll
    for (int p = tid; p < 1024; p += 128) {
      int kp = p >> 6;
      int c  = p & 63;
      int k  = 2 * kp;
      int kk = k0 + k;
      int nn = blockIdx.y * 64 + c;
      float v0 = 0.f, v1 = 0.f;
      if (kk < K && nn < N) {
        v0 = Wt[(size_t)kk * ldw + nn];
        v1 = Wt[(size_t)(kk + 1) * ldw + nn];
      }
      int dl = (c & 15) + 16 * ((k >> 3) & 1);
      *(v2h*)&sB[(c >> 4) * 512 + dl * 16 + ((k & 7) | ((k >> 4) << 3))] =
          (v2h){(_Float16)v0, (_Float16)v1};
    }
    __syncthreads();
    if (colW < N) {  // scalar branch; EXEC all-1s inside (WMMA requirement)
      v16h af = *(const v16h*)&sA[lane * 16];
      v16h bf = *(const v16h*)&sB[wave * 512 + lane * 16];
      acc = __builtin_amdgcn_wmma_f32_16x16x32_f16(false, af, false, bf,
                                                   (short)0, acc, false, false);
    }
    __syncthreads();
  }

  if (colW < N) {
    const int nn = colW + lr;
    const float bv = bias ? bias[nn] : 0.f;
    float beta = 0.f, ombeta = 1.f;
    if (skipX) {
      beta = *betaPtr;  // pre-sigmoided gate
      ombeta = 1.f - beta;
    }
#pragma unroll
    for (int r = 0; r < 8; ++r) {  // C/D layout: lane half selects M block of 8
      int mm = row0 + r + (half << 3);
      float o = acc[r] + bv;
      if (skipX) o = beta * o + ombeta * skipX[(size_t)mm * ldskip + nn];
      if (actY == 1) o = fmaxf(o, 0.f);
      else if (actY == 2) o = (o > 0.f) ? o : 0.01f * o;
      Y[(size_t)mm * ldy + nn] = o;
    }
  }
}

__global__ void fill_kernel(float* __restrict__ p, float v, long n) {
  long t = (long)blockIdx.x * blockDim.x + threadIdx.x;
  if (t < n) p[t] = v;
}

// alpha[e,h] = <krel[src,h,:], q[dst,h,:]> * prel[h] * scale; segment max via atomics
__global__ void edge_alpha_kernel(const float* __restrict__ krel, const float* __restrict__ q,
                                  const int* __restrict__ src, const int* __restrict__ dst,
                                  const float* __restrict__ prel, float scale,
                                  int H, int D, float* __restrict__ alpha,
                                  float* __restrict__ mbuf, long total) {
  long t = (long)blockIdx.x * blockDim.x + threadIdx.x;
  if (t >= total) return;
  int e = (int)(t / H), h = (int)(t % H);
  int sN = src[e], dN = dst[e];
  const float* kp = krel + (size_t)sN * H * D + h * D;
  const float* qp = q    + (size_t)dN * H * D + h * D;
  float a = 0.f;
  for (int f = 0; f < D; ++f) a += kp[f] * qp[f];
  a *= prel[h] * scale;
  alpha[t] = a;
  atomicMaxF(&mbuf[(size_t)dN * H + h], a);
}

__global__ void edge_exp_kernel(const int* __restrict__ dst, const float* __restrict__ mbuf,
                                float* __restrict__ alpha, float* __restrict__ sbuf,
                                int H, long total) {
  long t = (long)blockIdx.x * blockDim.x + threadIdx.x;
  if (t >= total) return;
  int e = (int)(t / H), h = (int)(t % H);
  float v = expf(alpha[t] - mbuf[(size_t)dst[e] * H + h]);
  alpha[t] = v;
  atomicAdd(&sbuf[(size_t)dst[e] * H + h], v);
}

// agg[dst,h,f] += vrel[src,h,f] * e/(s+eps); f32 atomic adds hit L2
__global__ void edge_msg_kernel(const int* __restrict__ src, const int* __restrict__ dst,
                                const float* __restrict__ vrel, const float* __restrict__ alpha,
                                const float* __restrict__ sbuf, float* __restrict__ agg,
                                int H, int D, long total) {
  long t = (long)blockIdx.x * blockDim.x + threadIdx.x;
  if (t >= total) return;
  int HD = H * D;
  int e = (int)(t / HD), c = (int)(t % HD);
  int h = c / D;
  int dN = dst[e];
  float a = alpha[(size_t)e * H + h] / (sbuf[(size_t)dN * H + h] + 1e-16f);
  atomicAdd(&agg[(size_t)dN * HD + c], vrel[(size_t)src[e] * HD + c] * a);
}

__global__ void gemv_sigmoid_kernel(const float* __restrict__ Xr, const float* __restrict__ w,
                                    const float* __restrict__ b, float* __restrict__ out, int n) {
  int i = blockIdx.x * blockDim.x + threadIdx.x;
  if (i >= n) return;
  float acc = b[0];
  const float* xp = Xr + (size_t)i * 64;
#pragma unroll
  for (int j = 0; j < 64; ++j) acc += xp[j] * w[j];
  out[i] = 1.f / (1.f + expf(-acc));
}

// ---------------------------------------------------------------------------
static inline void gemm(hipStream_t s, const float* X, int ldx, const float* Wt, int ldw,
                        const float* bias, float* Y, int ldy, int M, int K, int N,
                        int geluX = 0, int actY = 0,
                        const float* skipX = nullptr, int ldskip = 0,
                        const float* beta = nullptr) {
  dim3 g(M / 16, (N + 63) / 64);
  switch (K) {
    case 16:
      gemm_wmma_t<16><<<g, 128, 0, s>>>(X, ldx, Wt, ldw, bias, Y, ldy, M, N,
                                        geluX, actY, skipX, ldskip, beta);
      break;
    case 32:
      gemm_wmma_t<32><<<g, 128, 0, s>>>(X, ldx, Wt, ldw, bias, Y, ldy, M, N,
                                        geluX, actY, skipX, ldskip, beta);
      break;
    case 64:
      gemm_wmma_t<64><<<g, 128, 0, s>>>(X, ldx, Wt, ldw, bias, Y, ldy, M, N,
                                        geluX, actY, skipX, ldskip, beta);
      break;
    default:
      gemm_wmma_t<128><<<g, 128, 0, s>>>(X, ldx, Wt, ldw, bias, Y, ldy, M, N,
                                         geluX, actY, skipX, ldskip, beta);
      break;
  }
}

extern "C" void kernel_launch(void* const* d_in, const int* in_sizes, int n_in,
                              void* d_out, int out_size, void* d_ws, size_t ws_size,
                              hipStream_t stream) {
  (void)in_sizes; (void)n_in; (void)out_size; (void)ws_size;
  const float* x_c    = (const float*)d_in[0];
  const float* x_m    = (const float*)d_in[1];
  const int*   cm_src = (const int*)d_in[2];
  const int*   cm_dst = (const int*)d_in[3];
  const int*   mc_src = (const int*)d_in[4];
  const int*   mc_dst = (const int*)d_in[5];
  const float* P[60];
  for (int j = 0; j < 60; ++j) P[j] = (const float*)d_in[6 + j];
  // P layout (relative): 0..3 init_{c,m}.{w,b}
  // conv1 @4:  k_c.{w,b}=4,5 q_c=6,7 v_c=8,9 a_c=10,11 skip_c=12
  //            k_m=13,14 q_m=15,16 v_m=17,18 a_m=19,20 skip_m=21
  //            arel_cm=22 mrel_cm=23 prel_cm=24 arel_mc=25 mrel_mc=26 prel_mc=27
  // conv2 @28: same structure (k_c=28.. skip_m=45, arel_cm=46.. prel_mc=51)
  // proj0=52,53 proj1=54,55 cls0=56,57 cls1=58,59

  float* Wp = (float*)d_ws;
  size_t off = 0;
  auto take = [&](size_t nf) { float* p = Wp + off; off += nf; return p; };
  float* h_c   = take((size_t)NCN * 128);
  float* h_m   = take((size_t)NMN * 128);
  float* kC    = take((size_t)NCN * 128);
  float* qC    = take((size_t)NCN * 128);
  float* vC    = take((size_t)NCN * 128);
  float* kM    = take((size_t)NMN * 128);
  float* qM    = take((size_t)NMN * 128);
  float* vM    = take((size_t)NMN * 128);
  float* krel  = take((size_t)NCN * 128);
  float* vrel  = take((size_t)NCN * 128);
  float* ea    = take((size_t)NEE * 4);
  float* mb    = take((size_t)NCN * 4);
  float* sb    = take((size_t)NCN * 4);
  float* aggC  = take((size_t)NCN * 128);
  float* aggM  = take((size_t)NMN * 128);
  float* oC    = take((size_t)NCN * 128);
  float* oM    = take((size_t)NMN * 128);
  float* gates = take(2);  // sigmoid(skip_c), sigmoid(skip_m) for conv1
  // conv2 / head reuse (NM*128 == NC*64 exactly)
  float* kM2 = kM; float* qC2 = qC; float* vM2 = vM;
  float* zc  = oM; float* t1 = aggM; float* t2 = kC;

  auto fill = [&](float* p, float v, long n) {
    fill_kernel<<<(unsigned)((n + 255) / 256), 256, 0, stream>>>(p, v, n);
  };
  const long EH  = (long)NEE * 4;
  const int  TB  = 256;
  const float NEGINF = -3.0e38f;

  // ---- precompute sigmoid skip gates (conv1) ---------------------------
  prep_gates_kernel<<<1, 32, 0, stream>>>(P[12], P[21], gates);

  // ---- init linears: h = relu(x @ W + b) -------------------------------
  gemm(stream, x_c, 64, P[0], 128, P[1], h_c, 128, NCN, 64, 128, 0, 1);
  gemm(stream, x_m, 32, P[2], 128, P[3], h_m, 128, NMN, 32, 128, 0, 1);

  // ---- conv1 k/q/v projections -----------------------------------------
  gemm(stream, h_c, 128, P[4],  128, P[5],  kC, 128, NCN, 128, 128);
  gemm(stream, h_c, 128, P[6],  128, P[7],  qC, 128, NCN, 128, 128);
  gemm(stream, h_c, 128, P[8],  128, P[9],  vC, 128, NCN, 128, 128);
  gemm(stream, h_m, 128, P[13], 128, P[14], kM, 128, NMN, 128, 128);
  gemm(stream, h_m, 128, P[15], 128, P[16], qM, 128, NMN, 128, 128);
  gemm(stream, h_m, 128, P[17], 128, P[18], vM, 128, NMN, 128, 128);
  fill(aggC, 0.f, (long)NCN * 128);
  fill(aggM, 0.f, (long)NMN * 128);

  // ---- conv1 relation cm (src=c -> dst=m) ------------------------------
  for (int h = 0; h < 4; ++h) {  // per-head rel transforms hoisted to nodes
    gemm(stream, kC + h * 32, 128, P[22] + h * 1024, 32, nullptr, krel + h * 32, 128, NCN, 32, 32);
    gemm(stream, vC + h * 32, 128, P[23] + h * 1024, 32, nullptr, vrel + h * 32, 128, NCN, 32, 32);
  }
  fill(mb, NEGINF, (long)NMN * 4);
  fill(sb, 0.f,    (long)NMN * 4);
  edge_alpha_kernel<<<(unsigned)((EH + TB - 1) / TB), TB, 0, stream>>>(
      krel, qM, cm_src, cm_dst, P[24], 0.17677669529663687f, 4, 32, ea, mb, EH);
  edge_exp_kernel<<<(unsigned)((EH + TB - 1) / TB), TB, 0, stream>>>(cm_dst, mb, ea, sb, 4, EH);
  {
    long tot = (long)NEE * 128;
    edge_msg_kernel<<<(unsigned)((tot + TB - 1) / TB), TB, 0, stream>>>(
        cm_src, cm_dst, vrel, ea, sb, aggM, 4, 32, tot);
  }

  // ---- conv1 relation mc (src=m -> dst=c) ------------------------------
  for (int h = 0; h < 4; ++h) {
    gemm(stream, kM + h * 32, 128, P[25] + h * 1024, 32, nullptr, krel + h * 32, 128, NMN, 32, 32);
    gemm(stream, vM + h * 32, 128, P[26] + h * 1024, 32, nullptr, vrel + h * 32, 128, NMN, 32, 32);
  }
  fill(mb, NEGINF, (long)NCN * 4);
  fill(sb, 0.f,    (long)NCN * 4);
  edge_alpha_kernel<<<(unsigned)((EH + TB - 1) / TB), TB, 0, stream>>>(
      krel, qC, mc_src, mc_dst, P[27], 0.17677669529663687f, 4, 32, ea, mb, EH);
  edge_exp_kernel<<<(unsigned)((EH + TB - 1) / TB), TB, 0, stream>>>(mc_dst, mb, ea, sb, 4, EH);
  {
    long tot = (long)NEE * 128;
    edge_msg_kernel<<<(unsigned)((tot + TB - 1) / TB), TB, 0, stream>>>(
        mc_src, mc_dst, vrel, ea, sb, aggC, 4, 32, tot);
  }

  // ---- conv1 output: leaky( gate*lin(gelu(agg)) + (1-gate)*h ) ---------
  gemm(stream, aggC, 128, P[10], 128, P[11], oC, 128, NCN, 128, 128, 1, 2, h_c, 128, gates + 0);
  gemm(stream, aggM, 128, P[19], 128, P[20], oM, 128, NMN, 128, 128, 1, 2, h_m, 128, gates + 1);

  // ---- conv2 (only z_c needed -> only relation mc) ---------------------
  gemm(stream, oC, 128, P[30], 64, P[31], qC2, 64, NCN, 128, 64);  // q_c
  gemm(stream, oM, 128, P[37], 64, P[38], kM2, 64, NMN, 128, 64);  // k_m
  gemm(stream, oM, 128, P[41], 64, P[42], vM2, 64, NMN, 128, 64);  // v_m
  for (int h = 0; h < 4; ++h) {  // D=16 heads (K padded to 32 inside kernel)
    gemm(stream, kM2 + h * 16, 64, P[49] + h * 256, 16, nullptr, krel + h * 16, 64, NMN, 16, 16);
    gemm(stream, vM2 + h * 16, 64, P[50] + h * 256, 16, nullptr, vrel + h * 16, 64, NMN, 16, 16);
  }
  fill(aggC, 0.f,  (long)NCN * 64);
  fill(mb, NEGINF, (long)NCN * 4);
  fill(sb, 0.f,    (long)NCN * 4);
  edge_alpha_kernel<<<(unsigned)((EH + TB - 1) / TB), TB, 0, stream>>>(
      krel, qC2, mc_src, mc_dst, P[51], 0.25f, 4, 16, ea, mb, EH);
  edge_exp_kernel<<<(unsigned)((EH + TB - 1) / TB), TB, 0, stream>>>(mc_dst, mb, ea, sb, 4, EH);
  {
    long tot = (long)NEE * 64;
    edge_msg_kernel<<<(unsigned)((tot + TB - 1) / TB), TB, 0, stream>>>(
        mc_src, mc_dst, vrel, ea, sb, aggC, 4, 16, tot);
  }
  gemm(stream, aggC, 64, P[34], 64, P[35], zc, 64, NCN, 64, 64, 1, 0);  // z_c (no skip)

  // ---- heads ------------------------------------------------------------
  float* out_scores = (float*)d_out;
  float* out_proj   = (float*)d_out + NCN;
  gemm(stream, zc, 64, P[52], 64, P[53], t1, 64, NCN, 64, 64, 0, 1);        // proj0+relu
  gemm(stream, t1, 64, P[54], 64, P[55], out_proj, 64, NCN, 64, 64, 0, 0);  // proj1
  gemm(stream, zc, 64, P[56], 64, P[57], t2, 64, NCN, 64, 64, 0, 1);        // cls0+relu
  gemv_sigmoid_kernel<<<(NCN + 255) / 256, 256, 0, stream>>>(t2, P[58], P[59], out_scores, NCN);
}
